// GranMultiHeadAttention_25520695673474
// MI455X (gfx1250) — matrix operand order repaired and checked by reading, not compiled, not run on hardware
//
#include <hip/hip_runtime.h>
#include <hip/hip_bf16.h>

// ---------------------------------------------------------------------------
// GranMultiHeadAttention for MI455X (gfx1250), bf16 WMMA pipeline, v2.
// B=8, S=1024, EMB=512, H=8, DK=64, BH=64 (b*H+h), D_MODEL=512.
// v2: bf16 score/weight tensor, fused scores+softmax, 4-tile register
// blocking per wave (one A-fragment feeds 4 v_wmma), global prefetch.
// ---------------------------------------------------------------------------

typedef __attribute__((ext_vector_type(16))) __bf16 v16bf;
typedef __attribute__((ext_vector_type(8)))  float  v8f;

#define S_LEN 1024
#define EMB   512
#define NH    8
#define DK    64
#define BH    64
#define HD    512
#define NROW  8192   // B*S

// ---- WMMA helpers ---------------------------------------------------------

__device__ __forceinline__ v8f wmma_bf16(v16bf a, v16bf b, v8f c) {
    return __builtin_amdgcn_wmma_f32_16x16x32_bf16(
        /*neg_a=*/false, a, /*neg_b=*/false, b,
        /*c_mod=*/(short)0, c, /*reuse_a=*/false, /*reuse_b=*/false);
}

// A-matrix fragment: 16x32 (MxK), row-major source with stride lda.
// Lane layout: row = lane&15, half = lane>>4; elem j -> K=(j&7)+8*half+16*(j>>3).
template <typename T>
__device__ __forceinline__ v16bf load_A(const T* A, int lda, int kbase) {
    int lane = threadIdx.x & 31;
    int half = lane >> 4;
    int row  = lane & 15;
    const T* p = A + (size_t)row * lda + kbase + 8 * half;
    v16bf a;
#pragma unroll
    for (int j = 0; j < 8; ++j) a[j]     = (__bf16)(float)p[j];
#pragma unroll
    for (int j = 0; j < 8; ++j) a[8 + j] = (__bf16)(float)p[16 + j];
    return a;
}

// B-matrix fragment: 32x16 (KxN). Row n (=output column) holds K-contiguous
// contraction values, stride ldb. col = lane&15, elem j -> K=(j&7)+8*half+16*(j>>3).
template <typename T>
__device__ __forceinline__ v16bf load_B(const T* Bm, int ldb, int kbase) {
    int lane = threadIdx.x & 31;
    int half = lane >> 4;
    int col  = lane & 15;
    const T* p = Bm + (size_t)col * ldb + kbase + 8 * half;
    v16bf b;
#pragma unroll
    for (int j = 0; j < 8; ++j) b[j]     = (__bf16)(float)p[j];
#pragma unroll
    for (int j = 0; j < 8; ++j) b[8 + j] = (__bf16)(float)p[16 + j];
    return b;
}

// 16x64 strip: one shared A-fragment feeds 4 WMMA tiles per k-step.
template <typename TA, typename TB>
__device__ __forceinline__ void gemm_strip4(const TA* A, int lda,
                                            const TB* Bm, int ldb,
                                            size_t btile_stride, int K,
                                            v8f acc[4]) {
    for (int k = 0; k < K; k += 32) {
        v16bf a = load_A(A, lda, k);
#pragma unroll
        for (int t = 0; t < 4; ++t) {
            v16bf b = load_B(Bm + btile_stride * t, ldb, k);
            acc[t] = wmma_bf16(a, b, acc[t]);
        }
    }
}

// ---- Kernel 1: transpose+convert the four weight matrices to bf16 [o][e] --

__global__ void transpose_w4_kernel(const float* W0, const float* W1,
                                    const float* W2, const float* W3,
                                    __bf16* T0, __bf16* T1,
                                    __bf16* T2, __bf16* T3) {
    const float* W; __bf16* T;
    switch (blockIdx.y) {
        case 0:  W = W0; T = T0; break;
        case 1:  W = W1; T = T1; break;
        case 2:  W = W2; T = T2; break;
        default: W = W3; T = T3; break;
    }
    int idx = blockIdx.x * 256 + threadIdx.x;   // over 512*512
    int e = idx >> 9;
    int o = idx & 511;
    T[(size_t)o * 512 + e] = (__bf16)W[idx];
}

// ---- Kernel 2: fused QKV projections -------------------------------------
// grid (128, 2, 3), block 512: wave = (row-tile, col-group of 64).
// z=0: Q (scaled 1/8, [bh][s][d]); z=1: K ([bh][s][d]); z=2: V ([bh][d][s]).

__global__ void qkv_proj_kernel(const float* Xq, const float* Xk, const float* Xv,
                                const __bf16* Wqt, const __bf16* Wkt, const __bf16* Wvt,
                                const float* bq, const float* bk, const float* bv,
                                __bf16* Qs, __bf16* Kb, __bf16* Vt) {
    int mode = blockIdx.z;
    const float* X; const __bf16* Wt; const float* bias;
    if (mode == 0)      { X = Xq; Wt = Wqt; bias = bq; }
    else if (mode == 1) { X = Xk; Wt = Wkt; bias = bk; }
    else                { X = Xv; Wt = Wvt; bias = bv; }

    int w    = threadIdx.x >> 5;
    int lane = threadIdx.x & 31;
    int row0 = blockIdx.x * 64 + (w >> 2) * 16;
    int col0 = blockIdx.y * 256 + (w & 3) * 64;

    v8f acc[4] = {};
    gemm_strip4(X + (size_t)row0 * EMB, EMB,
                Wt + (size_t)col0 * EMB, EMB, (size_t)16 * EMB, EMB, acc);

    int half = lane >> 4;
    int c    = lane & 15;
#pragma unroll
    for (int t = 0; t < 4; ++t) {
        int col = col0 + t * 16 + c;
        int h = col >> 6, d = col & 63;
        float bb = bias[col];
#pragma unroll
        for (int i = 0; i < 8; ++i) {
            int row = row0 + i + 8 * half;
            int b = row >> 10, s = row & 1023;
            float v = acc[t][i] + bb;
            int bh = b * NH + h;
            if (mode == 0) {
                Qs[((size_t)bh * S_LEN + s) * DK + d] = (__bf16)(v * 0.125f);
            } else if (mode == 1) {
                Kb[((size_t)bh * S_LEN + s) * DK + d] = (__bf16)v;
            } else {
                Vt[((size_t)bh * DK + d) * S_LEN + s] = (__bf16)v;
            }
        }
    }
}

// ---- Kernel 3: edge-key bias, Pe[bh,q,k] = Qs[bh,q,:].Ek[q,k,:]  (bf16) --
// Per-q GEMM: A = Qs[:,q,:] (64x64, row stride S*DK), B = Ek[q] rows are
// d-contiguous = contraction-contiguous. grid (1024, 4), block 512:
// wave = (bh-tile, k-strip of 64). edges_key (fp32) read exactly once.

__global__ void edge_key_kernel(const __bf16* Qs, const float* Ek, __bf16* Pe) {
    int q    = blockIdx.x;
    int w    = threadIdx.x >> 5;
    int lane = threadIdx.x & 31;
    int bh0 = (w >> 2) * 16;
    int k0  = blockIdx.y * 256 + (w & 3) * 64;

    v8f acc[4] = {};
    gemm_strip4(Qs + ((size_t)bh0 * S_LEN + q) * DK, S_LEN * DK,
                Ek + ((size_t)q * S_LEN + k0) * DK, DK, (size_t)16 * DK, DK, acc);

    int half = lane >> 4;
    int c    = lane & 15;
#pragma unroll
    for (int t = 0; t < 4; ++t) {
#pragma unroll
        for (int i = 0; i < 8; ++i) {
            int bh = bh0 + i + 8 * half;
            int k  = k0 + t * 16 + c;
            Pe[((size_t)bh * S_LEN + q) * S_LEN + k] = (__bf16)acc[t][i];
        }
    }
}

// ---- Kernel 4: fused scores + bias + softmax -> bf16 weights (in place) --
// grid (64, 64): workgroup = (bh, q-tile of 16). Block 512: wave w owns
// k-columns [w*64, w*64+64). Row softmax: shfl_xor within 16-lane halves,
// then a 16x16 LDS stage across the 16 waves. Pe is read (edge bias) and
// overwritten (weights) by the same workgroup only -> race-free.

__global__ void scores_softmax_kernel(const __bf16* Qs, const __bf16* Kb,
                                      const float* attn_bias, __bf16* Pe) {
    int bh = blockIdx.x;
    int q0 = blockIdx.y * 16;
    int w    = threadIdx.x >> 5;
    int lane = threadIdx.x & 31;
    int half = lane >> 4;
    int c    = lane & 15;
    int k0   = w * 64;
    int b    = bh >> 3;

    v8f acc[4] = {};
    gemm_strip4(Qs + ((size_t)bh * S_LEN + q0) * DK, DK,
                Kb + ((size_t)bh * S_LEN + k0) * DK, DK, (size_t)16 * DK, DK, acc);

    // logits = QK^T + attn_bias + edge_bias
    float s[4][8];
#pragma unroll
    for (int t = 0; t < 4; ++t) {
#pragma unroll
        for (int i = 0; i < 8; ++i) {
            int q = q0 + i + 8 * half;
            int k = k0 + t * 16 + c;
            s[t][i] = acc[t][i]
                    + attn_bias[((size_t)b * S_LEN + q) * S_LEN + k]
                    + (float)Pe[((size_t)bh * S_LEN + q) * S_LEN + k];
        }
    }

    __shared__ float red[16][16];   // [row-in-tile][wave]

    // --- row max ---
    float m[8];
#pragma unroll
    for (int i = 0; i < 8; ++i) {
        m[i] = s[0][i];
#pragma unroll
        for (int t = 1; t < 4; ++t) m[i] = fmaxf(m[i], s[t][i]);
#pragma unroll
        for (int off = 8; off >= 1; off >>= 1)
            m[i] = fmaxf(m[i], __shfl_xor(m[i], off, 32));
    }
    if (c == 0) {
#pragma unroll
        for (int i = 0; i < 8; ++i) red[i + 8 * half][w] = m[i];
    }
    __syncthreads();
    float rmax[8];
#pragma unroll
    for (int i = 0; i < 8; ++i) {
        int r = i + 8 * half;
        float mm = red[r][0];
#pragma unroll
        for (int ww = 1; ww < 16; ++ww) mm = fmaxf(mm, red[r][ww]);
        rmax[i] = mm;
    }
    __syncthreads();

    // --- exp + row sum ---
    float sum[8];
#pragma unroll
    for (int i = 0; i < 8; ++i) {
        sum[i] = 0.f;
#pragma unroll
        for (int t = 0; t < 4; ++t) {
            s[t][i] = __expf(s[t][i] - rmax[i]);
            sum[i] += s[t][i];
        }
#pragma unroll
        for (int off = 8; off >= 1; off >>= 1)
            sum[i] += __shfl_xor(sum[i], off, 32);
    }
    if (c == 0) {
#pragma unroll
        for (int i = 0; i < 8; ++i) red[i + 8 * half][w] = sum[i];
    }
    __syncthreads();
    float rinv[8];
#pragma unroll
    for (int i = 0; i < 8; ++i) {
        int r = i + 8 * half;
        float ss = red[r][0];
#pragma unroll
        for (int ww = 1; ww < 16; ++ww) ss += red[r][ww];
        rinv[i] = 1.0f / ss;
    }

    // --- write normalized weights (bf16, overwrite Pe) ---
#pragma unroll
    for (int t = 0; t < 4; ++t) {
#pragma unroll
        for (int i = 0; i < 8; ++i) {
            int q = q0 + i + 8 * half;
            int k = k0 + t * 16 + c;
            Pe[((size_t)bh * S_LEN + q) * S_LEN + k] = (__bf16)(s[t][i] * rinv[i]);
        }
    }
}

// ---- Kernel 5: edge-value, OE[bh,q,d] = sum_k W[bh,q,k]*Ev[q,k,d] --------
// Per-q GEMM, contraction k=1024. Ev is d-contiguous -> stage 32k x 64d tile
// transposed into LDS. grid 1024, block 128 (4 waves = 4 bh-tiles), each wave
// computes 16bh x 64d with a shared A-fragment. edges_value read once;
// next tile prefetched (global_prefetch_b8).

__global__ void edge_value_kernel(const __bf16* Pe, const float* Ev, float* OE) {
    int q    = blockIdx.x;
    int tid  = threadIdx.x;          // 0..127
    int w    = tid >> 5;             // bh-tile
    int lane = tid & 31;
    int bh0  = w * 16;

    __shared__ __bf16 Bs[64 * 36];   // [d][k] transposed tile, padded rows

    const __bf16* Arow = Pe + ((size_t)bh0 * S_LEN + q) * S_LEN;
    const float*  Evq  = Ev + (size_t)q * S_LEN * DK;

    v8f acc[4] = {};
    for (int kk0 = 0; kk0 < S_LEN; kk0 += 32) {
        __syncthreads();
#pragma unroll
        for (int i = 0; i < 16; ++i) {
            int t  = tid + i * 128;          // 0..2047
            int kk = t >> 6;                 // 0..31
            int d  = t & 63;                 // 0..63
            Bs[d * 36 + kk] = (__bf16)Evq[(size_t)(kk0 + kk) * DK + d];
        }
        if (kk0 + 32 < S_LEN) {
            __builtin_prefetch(Evq + (size_t)(kk0 + 32) * DK + tid * 16, 0, 1);
        }
        __syncthreads();
        v16bf a = load_A(Arow + kk0, S_LEN * S_LEN, 0);
#pragma unroll
        for (int t = 0; t < 4; ++t) {
            v16bf bfr = load_B(Bs + (t * 16) * 36, 36, 0);
            acc[t] = wmma_bf16(a, bfr, acc[t]);
        }
    }

    int half = lane >> 4;
    int c    = lane & 15;
#pragma unroll
    for (int t = 0; t < 4; ++t) {
#pragma unroll
        for (int i = 0; i < 8; ++i) {
            int bh = bh0 + i + 8 * half;
            OE[((size_t)bh * S_LEN + q) * DK + t * 16 + c] = acc[t][i];
        }
    }
}

// ---- Kernel 6: attn out = W @ V + OE, merged-head store ------------------
// grid (64, 4), block 512: per block one bh, 256 q rows; wave w owns 16 q
// rows x all 64 d (4 tiles, shared A-fragment). K = 1024 (bf16 weights).

__global__ void attn_out_kernel(const __bf16* Pe, const __bf16* Vt,
                                const float* OE, __bf16* Ob) {
    int bh   = blockIdx.x;
    int w    = threadIdx.x >> 5;
    int lane = threadIdx.x & 31;
    int q0   = blockIdx.y * 256 + w * 16;

    v8f acc[4] = {};
    gemm_strip4(Pe + ((size_t)bh * S_LEN + q0) * S_LEN, S_LEN,
                Vt + (size_t)bh * DK * S_LEN, S_LEN, (size_t)16 * S_LEN,
                S_LEN, acc);

    int half = lane >> 4;
    int c    = lane & 15;
    int b = bh >> 3, h = bh & 7;
#pragma unroll
    for (int t = 0; t < 4; ++t) {
#pragma unroll
        for (int i = 0; i < 8; ++i) {
            int q = q0 + i + 8 * half;
            int d = t * 16 + c;
            float v = acc[t][i] + OE[((size_t)bh * S_LEN + q) * DK + d];
            Ob[((size_t)b * S_LEN + q) * HD + h * DK + d] = (__bf16)v;
        }
    }
}

// ---- Kernel 7: output projection out = Ob @ Wp + bp ----------------------
// grid (128, 2), block 512: wave = (row-tile, col-strip of 64), K = 512.

__global__ void out_proj_kernel(const __bf16* Ob, const __bf16* Wpt,
                                const float* bp, float* out) {
    int w    = threadIdx.x >> 5;
    int lane = threadIdx.x & 31;
    int row0 = blockIdx.x * 64 + (w >> 2) * 16;
    int col0 = blockIdx.y * 256 + (w & 3) * 64;

    v8f acc[4] = {};
    gemm_strip4(Ob  + (size_t)row0 * HD, HD,
                Wpt + (size_t)col0 * HD, HD, (size_t)16 * HD, HD, acc);

    int half = lane >> 4;
    int c    = lane & 15;
#pragma unroll
    for (int t = 0; t < 4; ++t) {
#pragma unroll
        for (int i = 0; i < 8; ++i) {
            int row = row0 + i + 8 * half;
            int col = col0 + t * 16 + c;
            out[(size_t)row * HD + col] = acc[t][i] + bp[col];
        }
    }
}

// ---------------------------------------------------------------------------

extern "C" void kernel_launch(void* const* d_in, const int* in_sizes, int n_in,
                              void* d_out, int out_size, void* d_ws, size_t ws_size,
                              hipStream_t stream) {
    const float* queries = (const float*)d_in[0];
    const float* keys    = (const float*)d_in[1];
    const float* values  = (const float*)d_in[2];
    const float* Ek      = (const float*)d_in[3];
    const float* Ev      = (const float*)d_in[4];
    const float* abias   = (const float*)d_in[5];
    const float* Wq = (const float*)d_in[6];
    const float* bq = (const float*)d_in[7];
    const float* Wk = (const float*)d_in[8];
    const float* bk = (const float*)d_in[9];
    const float* Wv = (const float*)d_in[10];
    const float* bv = (const float*)d_in[11];
    const float* Wp = (const float*)d_in[12];
    const float* bp = (const float*)d_in[13];
    float* out = (float*)d_out;

    char* ws = (char*)d_ws;
    size_t off = 0;
    auto alloc = [&](size_t bytes) -> void* {
        void* p = ws + off;
        off += (bytes + 255) & ~(size_t)255;
        return p;
    };

    __bf16* Wqt = (__bf16*)alloc((size_t)EMB * HD * 2);
    __bf16* Wkt = (__bf16*)alloc((size_t)EMB * HD * 2);
    __bf16* Wvt = (__bf16*)alloc((size_t)EMB * HD * 2);
    __bf16* Wpt = (__bf16*)alloc((size_t)HD * HD * 2);
    __bf16* Qs  = (__bf16*)alloc((size_t)BH * S_LEN * DK * 2);
    __bf16* Kb  = (__bf16*)alloc((size_t)BH * S_LEN * DK * 2);
    __bf16* Vt  = (__bf16*)alloc((size_t)BH * DK * S_LEN * 2);
    __bf16* Pe  = (__bf16*)alloc((size_t)BH * S_LEN * S_LEN * 2); // edge bias -> weights
    float*  OE  = (float*) alloc((size_t)BH * S_LEN * DK * 4);
    __bf16* Ob  = (__bf16*)alloc((size_t)NROW * HD * 2);
    (void)ws_size; (void)in_sizes; (void)n_in; (void)out_size;

    // 1. weight transpose -> bf16 [o][e]
    transpose_w4_kernel<<<dim3(512 * 512 / 256, 4), 256, 0, stream>>>(
        Wq, Wk, Wv, Wp, Wqt, Wkt, Wvt, Wpt);

    // 2. QKV projections (Q pre-scaled, V stored transposed)
    qkv_proj_kernel<<<dim3(NROW / 64, HD / 256, 3), 512, 0, stream>>>(
        queries, keys, values, Wqt, Wkt, Wvt, bq, bk, bv, Qs, Kb, Vt);

    // 3. edge-key bias -> Pe (bf16), edges_key read once
    edge_key_kernel<<<dim3(S_LEN, S_LEN / 256), 512, 0, stream>>>(Qs, Ek, Pe);

    // 4. fused scores + attn_bias + edge bias + softmax -> bf16 weights in Pe
    scores_softmax_kernel<<<dim3(BH, S_LEN / 16), 512, 0, stream>>>(
        Qs, Kb, abias, Pe);

    // 5. edge-value contribution (edges_value read once)
    edge_value_kernel<<<dim3(S_LEN), 128, 0, stream>>>(Pe, Ev, OE);

    // 6. W @ V + edge-out, merged-head bf16 store
    attn_out_kernel<<<dim3(BH, S_LEN / 256), 512, 0, stream>>>(Pe, Vt, OE, Ob);

    // 7. output projection
    out_proj_kernel<<<dim3(NROW / 64, HD / 256), 512, 0, stream>>>(Ob, Wpt, bp, out);
}
